// DualInputSFMModel_38878043963677
// MI455X (gfx1250) — matrix operand order repaired and checked by reading, not compiled, not run on hardware
//
#include <hip/hip_runtime.h>
#include <hip/hip_bf16.h>
#include <math.h>

#define Bn   256
#define Tn   128
#define SIGn 64
#define METn 32
#define Hn   256
#define Fn   16
#define NCAT 1040   // 4*H + F   (gate cols: i[0,256) ste[256,512) c[512,768) o[768,1024) fre[1024,1040))
#define KCAT 352    // H + SIG + MET
#define TWO_PI_F 6.283185307179586f

typedef __attribute__((ext_vector_type(16))) _Float16 v16h;
typedef __attribute__((ext_vector_type(8)))  float    v8f;

union Frag16 { v16h v; unsigned int u[8]; };

// ---------------------------------------------------------------------------
// Init kernels
// ---------------------------------------------------------------------------
__global__ void zero_f32_kernel(float* p, int n) {
    int i = blockIdx.x * blockDim.x + threadIdx.x;
    if (i < n) p[i] = 0.0f;
}

__global__ void zero_f16_kernel(_Float16* p, int n) {
    int i = blockIdx.x * blockDim.x + threadIdx.x;
    if (i < n) p[i] = (_Float16)0.0f;
}

// Wcat^T in f16:  wcat[j][k],  j in [0,1040), k in [0,352)
__global__ void build_wcat_kernel(
    const float* Ui, const float* Uste, const float* Uc, const float* Uo, const float* Ufre,
    const float* Wis, const float* Wstes, const float* Wcs, const float* Wos, const float* Wfres,
    const float* Wim, const float* Wstem, const float* Wcm, const float* Wom, const float* Wfrem,
    _Float16* wcat)
{
    int idx = blockIdx.x * blockDim.x + threadIdx.x;
    if (idx >= NCAT * KCAT) return;
    int j = idx / KCAT, k = idx % KCAT;
    int g, jl, ld;
    if (j < 1024) { g = j >> 8; jl = j & 255; ld = Hn; }
    else          { g = 4; jl = j - 1024; ld = Fn; }
    const float* U[5]  = {Ui,  Uste,  Uc,  Uo,  Ufre};
    const float* Ws[5] = {Wis, Wstes, Wcs, Wos, Wfres};
    const float* Wm[5] = {Wim, Wstem, Wcm, Wom, Wfrem};
    float v;
    if (k < Hn)              v = U[g][k * ld + jl];
    else if (k < Hn + SIGn)  v = Ws[g][(k - Hn) * ld + jl];
    else                     v = Wm[g][(k - Hn - SIGn) * ld + jl];
    wcat[(size_t)j * KCAT + k] = (_Float16)v;
}

__global__ void build_bias_kernel(const float* bi, const float* bste, const float* bc,
                                  const float* bo, const float* bfre, float* biasc)
{
    int j = blockIdx.x * blockDim.x + threadIdx.x;
    if (j >= NCAT) return;
    float v;
    if (j < 256)       v = bi[j];
    else if (j < 512)  v = bste[j - 256];
    else if (j < 768)  v = bc[j - 512];
    else if (j < 1024) v = bo[j - 768];
    else               v = bfre[j - 1024];
    biasc[j] = v;
}

// Repack sig (B,T,64) + met (B,T,32) -> sm[t][b][96] in f16
__global__ void pack_sm_kernel(const float* sig, const float* met, _Float16* sm)
{
    int idx = blockIdx.x * blockDim.x + threadIdx.x;
    if (idx >= Bn * Tn * 96) return;
    int b = idx / (Tn * 96);
    int r = idx % (Tn * 96);
    int t = r / 96, c = r % 96;
    float v = (c < SIGn) ? sig[((size_t)b * Tn + t) * SIGn + c]
                         : met[((size_t)b * Tn + t) * METn + (c - SIGn)];
    sm[((size_t)t * Bn + b) * 96 + c] = (_Float16)v;
}

// trig[t][f][2] = {cos, sin} of 2*pi*(t+1)*(f/F)
__global__ void trig_kernel(float* trig)
{
    int i = blockIdx.x * blockDim.x + threadIdx.x;
    if (i >= Tn * Fn) return;
    int t = i >> 4, f = i & 15;
    float omega = TWO_PI_F * (float)(t + 1) * ((float)f / (float)Fn);
    float s, c;
    sincosf(omega, &s, &c);
    trig[2 * i]     = c;
    trig[2 * i + 1] = s;
}

// ---------------------------------------------------------------------------
// Per-step fused gate GEMM:  gact[b][j] = act( [h|xs|xm](b,:) @ wcat^T(j,:) + bias[j] )
// One wave32 computes one 16x16 tile via v_wmma_f32_16x16x32_f16, K=352 (11 k-tiles).
// Tiles: 16 (M) x 65 (N) = 1040 = 130 blocks * 8 waves. EXEC all-ones everywhere.
// ---------------------------------------------------------------------------
__global__ __launch_bounds__(256)
void gates_gemm_kernel(const _Float16* __restrict__ hbuf,  // [B][256] f16
                       const _Float16* __restrict__ smt,   // [B][96]  f16 (already offset by t)
                       const _Float16* __restrict__ wcat,  // [1040][352] f16
                       const float*    __restrict__ biasc, // [1040]
                       float*          __restrict__ gact)  // [B][1040]
{
    const int lane = threadIdx.x & 31;
    const int wave = threadIdx.x >> 5;
    const int tile = blockIdx.x * 8 + wave;          // 0..1039
    const int m0   = (tile & 15) << 4;               // row tile
    const int n0   = (tile >> 4) << 4;               // col tile (0..1024)
    const int l15  = lane & 15;
    const int grp  = lane >> 4;
    const int row  = m0 + l15;                       // A row this lane loads
    const int ncol = n0 + l15;                       // B col / C col

    float bias = biasc[ncol];
    v8f acc;
#pragma unroll
    for (int r = 0; r < 8; ++r) acc[r] = bias;

#pragma unroll
    for (int kt = 0; kt < 11; ++kt) {
        const _Float16* abase;
        int astride, acol0;
        if (kt < 8) { abase = hbuf; astride = Hn; acol0 = kt * 32; }        // h part
        else        { abase = smt;  astride = 96; acol0 = kt * 32 - Hn; }   // [xs|xm] part
        Frag16 a, b;
#pragma unroll
        for (int v = 0; v < 8; ++v) {
            // A 16x32 f16 layout (ISA 7.12.2): VGPR v holds K pair; lanes16-31 shift K by 8 (v<4) / by 8 (v>=4)
            int ka = ((v & 3) << 1) + ((v >= 4) ? 16 : 0) + (grp << 3);
            a.u[v] = *(const unsigned int*)(abase + (size_t)row * astride + acol0 + ka);
            // B 32x16 f16 layout: same K order as A (VGPR v holds K pair; lanes16-31 shift K by 8)
            int kb = ((v & 3) << 1) + ((v >= 4) ? 16 : 0) + (grp << 3);
            b.u[v] = *(const unsigned int*)(wcat + (size_t)ncol * KCAT + kt * 32 + kb);
        }
        acc = __builtin_amdgcn_wmma_f32_16x16x32_f16(
            false, a.v, false, b.v, (short)0, acc, false, false);
    }

    // c-gate columns [512,768) get tanh; everything else hard-sigmoid.
    // Region boundaries are multiples of 16 -> uniform per tile.
    const bool is_tanh = (n0 >= 512) && (n0 < 768);
#pragma unroll
    for (int r = 0; r < 8; ++r) {
        int mrow = m0 + (grp << 3) + r;              // C/D layout: M = r + 8*grp
        float v = acc[r];
        v = is_tanh ? tanhf(v)
                    : fminf(fmaxf(v * (1.0f / 6.0f) + 0.5f, 0.0f), 1.0f);
        gact[(size_t)mrow * NCAT + ncol] = v;
    }
}

// ---------------------------------------------------------------------------
// Per-step state update: one thread per (b,h); F=16 inner loop.
// S stays L2-resident (8 MB << 192 MB).
// ---------------------------------------------------------------------------
__global__ __launch_bounds__(256)
void state_update_kernel(const float* __restrict__ gact,
                         float* __restrict__ S_re, float* __restrict__ S_im,
                         const float* __restrict__ trig_t,  // [F][2] for this step
                         const float* __restrict__ Ua,      // [F]
                         const float* __restrict__ ba,      // [H]
                         float* __restrict__ h32, _Float16* __restrict__ h16)
{
    int idx = blockIdx.x * blockDim.x + threadIdx.x;  // b*H + h
    int b = idx >> 8, hh = idx & 255;
    const float* g = gact + (size_t)b * NCAT;
    float gi = g[hh], gste = g[256 + hh], gc = g[512 + hh], go = g[768 + hh];
    float c = gi * gc;
    float acc = 0.0f;
    float* sre = S_re + (size_t)idx * Fn;
    float* sim = S_im + (size_t)idx * Fn;
#pragma unroll
    for (int f = 0; f < Fn; ++f) {
        float fg = gste * g[1024 + f];
        float re = fg * sre[f] + c * trig_t[2 * f];
        float im = fg * sim[f] + c * trig_t[2 * f + 1];
        sre[f] = re;
        sim[f] = im;
        acc += (re * re + im * im) * Ua[f];
    }
    float a  = tanhf(acc + ba[hh]);
    float hv = go * a;
    h32[idx] = hv;
    h16[idx] = (_Float16)hv;   // feeds next step's GEMM A-matrix
}

// out[b] = (h[b,:] @ W_p + b_p) * fc_w + fc_b   (OUT==1)
__global__ __launch_bounds__(256)
void final_proj_kernel(const float* __restrict__ h32, const float* __restrict__ Wp,
                       const float* __restrict__ bp, const float* __restrict__ fcw,
                       const float* __restrict__ fcb, float* __restrict__ out)
{
    __shared__ float red[256];
    int b = blockIdx.x, tid = threadIdx.x;
    red[tid] = h32[(size_t)b * Hn + tid] * Wp[tid];
    __syncthreads();
    for (int s = 128; s > 0; s >>= 1) {
        if (tid < s) red[tid] += red[tid + s];
        __syncthreads();
    }
    if (tid == 0) out[b] = (red[0] + bp[0]) * fcw[0] + fcb[0];
}

// ---------------------------------------------------------------------------
static inline size_t align256(size_t x) { return (x + 255) & ~(size_t)255; }

extern "C" void kernel_launch(void* const* d_in, const int* in_sizes, int n_in,
                              void* d_out, int out_size, void* d_ws, size_t ws_size,
                              hipStream_t stream) {
    (void)in_sizes; (void)n_in; (void)out_size; (void)ws_size;
    const float* sig   = (const float*)d_in[0];
    const float* met   = (const float*)d_in[1];
    const float* Wis   = (const float*)d_in[2];
    const float* Wstes = (const float*)d_in[3];
    const float* Wfres = (const float*)d_in[4];
    const float* Wcs   = (const float*)d_in[5];
    const float* Wos   = (const float*)d_in[6];
    const float* Wim   = (const float*)d_in[7];
    const float* Wstem = (const float*)d_in[8];
    const float* Wfrem = (const float*)d_in[9];
    const float* Wcm   = (const float*)d_in[10];
    const float* Wom   = (const float*)d_in[11];
    const float* Ui    = (const float*)d_in[12];
    const float* bi    = (const float*)d_in[13];
    const float* Uste  = (const float*)d_in[14];
    const float* bste  = (const float*)d_in[15];
    const float* Ufre  = (const float*)d_in[16];
    const float* bfre  = (const float*)d_in[17];
    const float* Uc    = (const float*)d_in[18];
    const float* bc    = (const float*)d_in[19];
    const float* Uo    = (const float*)d_in[20];
    const float* bo    = (const float*)d_in[21];
    const float* Ua    = (const float*)d_in[22];
    const float* ba    = (const float*)d_in[23];
    const float* Wp    = (const float*)d_in[24];
    const float* bp    = (const float*)d_in[25];
    const float* fcw   = (const float*)d_in[26];
    const float* fcb   = (const float*)d_in[27];
    float* out = (float*)d_out;

    // Workspace layout
    char* w = (char*)d_ws;
    size_t o = 0;
    float*    S_re  = (float*)(w + o);    o += (size_t)Bn * Hn * Fn * 4;   // 4 MiB
    float*    S_im  = (float*)(w + o);    o += (size_t)Bn * Hn * Fn * 4;   // 4 MiB (contiguous after S_re)
    float*    h32   = (float*)(w + o);    o += (size_t)Bn * Hn * 4;
    _Float16* h16   = (_Float16*)(w + o); o += align256((size_t)Bn * Hn * 2);
    _Float16* sm    = (_Float16*)(w + o); o += align256((size_t)Tn * Bn * 96 * 2);
    _Float16* wcat  = (_Float16*)(w + o); o += align256((size_t)NCAT * KCAT * 2);
    float*    biasc = (float*)(w + o);    o += align256((size_t)NCAT * 4);
    float*    gact  = (float*)(w + o);    o += align256((size_t)Bn * NCAT * 4);
    float*    trig  = (float*)(w + o);    o += align256((size_t)Tn * Fn * 2 * 4);

    // ---- init (recomputed each call; deterministic) ----
    {
        int nS = 2 * Bn * Hn * Fn;   // S_re + S_im contiguous
        zero_f32_kernel<<<(nS + 255) / 256, 256, 0, stream>>>(S_re, nS);
        zero_f16_kernel<<<(Bn * Hn + 255) / 256, 256, 0, stream>>>(h16, Bn * Hn);
        int nW = NCAT * KCAT;
        build_wcat_kernel<<<(nW + 255) / 256, 256, 0, stream>>>(
            Ui, Uste, Uc, Uo, Ufre, Wis, Wstes, Wcs, Wos, Wfres,
            Wim, Wstem, Wcm, Wom, Wfrem, wcat);
        build_bias_kernel<<<(NCAT + 255) / 256, 256, 0, stream>>>(bi, bste, bc, bo, bfre, biasc);
        int nSM = Bn * Tn * 96;
        pack_sm_kernel<<<(nSM + 255) / 256, 256, 0, stream>>>(sig, met, sm);
        trig_kernel<<<(Tn * Fn + 255) / 256, 256, 0, stream>>>(trig);
    }

    // ---- sequential scan over T ----
    for (int t = 0; t < Tn; ++t) {
        gates_gemm_kernel<<<130, 256, 0, stream>>>(
            h16, sm + (size_t)t * Bn * 96, wcat, biasc, gact);
        state_update_kernel<<<(Bn * Hn) / 256, 256, 0, stream>>>(
            gact, S_re, S_im, trig + (size_t)t * Fn * 2, Ua, ba, h32, h16);
    }

    // ---- final projection ----
    final_proj_kernel<<<Bn, 256, 0, stream>>>(h32, Wp, bp, fcw, fcb, out);
}
